// DGCNN_71837622993136
// MI455X (gfx1250) — matrix operand order repaired and checked by reading, not compiled, not run on hardware
//
#include <hip/hip_runtime.h>
#include <hip/hip_bf16.h>

typedef _Float16 v16h __attribute__((ext_vector_type(16)));
typedef float    v8f  __attribute__((ext_vector_type(8)));

#define HID 32
#define TOPK 20
#define NPG 512   // nodes per graph

// ---------------------------------------------------------------------------
// Utility kernels
// ---------------------------------------------------------------------------
__global__ void k_zero(float* __restrict__ p, int n) {
    int i = blockIdx.x * blockDim.x + threadIdx.x;
    if (i < n) p[i] = 0.0f;
}

__global__ void k_deg(const int* __restrict__ dst, float* __restrict__ deg, int E) {
    int e = blockIdx.x * blockDim.x + threadIdx.x;
    if (e < E) atomicAdd(&deg[dst[e]], 1.0f);
}

__global__ void k_dis(float* __restrict__ d, int n) {
    int i = blockIdx.x * blockDim.x + threadIdx.x;
    if (i < n) d[i] = rsqrtf(d[i] + 1.0f);   // +1 self loop
}

// ---------------------------------------------------------------------------
// WMMA GEMM: Y[NN,32] = X[NN,K] @ W[K,32]   (K multiple of 32)
// One wave per 16-row tile; two 16-col accumulator tiles (N=32).
// f32 -> f16 convert in registers, f32 accumulate via v_wmma_f32_16x16x32_f16.
// ---------------------------------------------------------------------------
__global__ void k_gemm_wmma_n32(const float* __restrict__ X,
                                const float* __restrict__ W,
                                float* __restrict__ Y,
                                int K, int nTiles) {
    const int lane = threadIdx.x & 31;
    const int wave = threadIdx.x >> 5;
    const int tile = blockIdx.x * (blockDim.x >> 5) + wave;
    if (tile >= nTiles) return;
    const int row0 = tile * 16;
    const int m  = lane & 15;   // A row / B col / D col within tile
    const int hi = lane >> 4;

    v8f acc0 = {};
    v8f acc1 = {};

    for (int ks = 0; ks < K; ks += 32) {
        v16h a, b0, b1;
        const float* xr = X + (size_t)(row0 + m) * K + ks;
#pragma unroll
        for (int i = 0; i < 16; ++i) {
            // A: 16x32, lane = row, K contiguous per half-wave
            int ka = 16 * hi + i;
            a[i] = (_Float16)xr[ka];
            // B: 32x16, lane = column, K contiguous per half-wave
            int kb = ks + 16 * hi + i;
            b0[i] = (_Float16)W[kb * HID + m];
            b1[i] = (_Float16)W[kb * HID + 16 + m];
        }
        acc0 = __builtin_amdgcn_wmma_f32_16x16x32_f16(false, a, false, b0,
                                                      (short)0, acc0, false, false);
        acc1 = __builtin_amdgcn_wmma_f32_16x16x32_f16(false, a, false, b1,
                                                      (short)0, acc1, false, false);
    }

#pragma unroll
    for (int v = 0; v < 8; ++v) {
        int orow = row0 + v + 8 * hi;       // C/D layout: VGPR v -> M = v + 8*hi
        Y[(size_t)orow * HID + m]      = acc0[v];
        Y[(size_t)orow * HID + 16 + m] = acc1[v];
    }
}

// ---------------------------------------------------------------------------
// GCN propagation (32-channel layers)
// ---------------------------------------------------------------------------
// self-loop init: agg[i,c] = h[i,c] * dis[i]^2
__global__ void k_init_self32(const float* __restrict__ h, const float* __restrict__ dis,
                              float* __restrict__ agg, int nn) {
    int idx = blockIdx.x * blockDim.x + threadIdx.x;
    if (idx >= nn * HID) return;
    int i = idx >> 5;
    float d = dis[i];
    agg[idx] = h[idx] * d * d;
}

// one wave per edge, one lane per channel (wave32 == 32 channels)
__global__ void k_edge_scatter32(const int* __restrict__ src, const int* __restrict__ dst,
                                 const float* __restrict__ h, const float* __restrict__ dis,
                                 float* __restrict__ agg, int E) {
    int e = blockIdx.x * (blockDim.x >> 5) + (threadIdx.x >> 5);
    int lane = threadIdx.x & 31;
    if (e >= E) return;
    int s = src[e], d = dst[e];
    float nrm = dis[s] * dis[d];
    atomicAdd(&agg[(size_t)d * HID + lane], h[(size_t)s * HID + lane] * nrm);
}

__global__ void k_fin_tanh32(float* __restrict__ agg, const float* __restrict__ b, int nn) {
    int idx = blockIdx.x * blockDim.x + threadIdx.x;
    if (idx >= nn * HID) return;
    agg[idx] = tanhf(agg[idx] + b[idx & 31]);
}

// ---------------------------------------------------------------------------
// Layer 3 (HID -> 1)
// ---------------------------------------------------------------------------
__global__ void k_node_dot32(const float* __restrict__ x2, const float* __restrict__ W2,
                             float* __restrict__ h3, int nn) {
    int i = blockIdx.x * blockDim.x + threadIdx.x;
    if (i >= nn) return;
    float s = 0.0f;
#pragma unroll
    for (int c = 0; c < HID; ++c) s += x2[(size_t)i * HID + c] * W2[c];
    h3[i] = s;
}

__global__ void k_init_self1(const float* __restrict__ h3, const float* __restrict__ dis,
                             float* __restrict__ x3, int nn) {
    int i = blockIdx.x * blockDim.x + threadIdx.x;
    if (i >= nn) return;
    float d = dis[i];
    x3[i] = h3[i] * d * d;
}

__global__ void k_edge_scatter1(const int* __restrict__ src, const int* __restrict__ dst,
                                const float* __restrict__ h3, const float* __restrict__ dis,
                                float* __restrict__ x3, int E) {
    int e = blockIdx.x * blockDim.x + threadIdx.x;
    if (e >= E) return;
    int s = src[e], d = dst[e];
    atomicAdd(&x3[d], h3[s] * dis[s] * dis[d]);
}

__global__ void k_fin_tanh1(float* __restrict__ x3, const float* __restrict__ b2, int nn) {
    int i = blockIdx.x * blockDim.x + threadIdx.x;
    if (i >= nn) return;
    x3[i] = tanhf(x3[i] + b2[0]);
}

// ---------------------------------------------------------------------------
// global_sort_pool: per-graph stable descending rank by x3; keep top-20 nodes.
// One block (NPG threads) per graph; rank via counting (matches stable argsort).
// ---------------------------------------------------------------------------
__global__ void k_sortpool(const float* __restrict__ x3, int* __restrict__ topk) {
    __shared__ float v[NPG];
    int g = blockIdx.x, j = threadIdx.x;
    int base = g * NPG;
    v[j] = x3[base + j];
    __syncthreads();
    float vj = v[j];
    int rank = 0;
    for (int k = 0; k < NPG; ++k) {
        float vk = v[k];
        rank += (vk > vj) || (vk == vj && k < j);
    }
    if (rank < TOPK) topk[g * TOPK + rank] = base + j;
}

// ---------------------------------------------------------------------------
// Tail: gather pooled [20,65] -> conv1(1->16,k=65,s=65) -> relu -> maxpool(2)
//       -> conv2(16->32,k=5) -> relu -> flatten(192) -> lin1(128)+relu -> lin2(2)
// One block per graph, everything in LDS.
// ---------------------------------------------------------------------------
__global__ void k_tail(const float* __restrict__ x1, const float* __restrict__ x2,
                       const float* __restrict__ x3, const int* __restrict__ topk,
                       const float* __restrict__ c1w, const float* __restrict__ c1b,
                       const float* __restrict__ c2w, const float* __restrict__ c2b,
                       const float* __restrict__ l1w, const float* __restrict__ l1b,
                       const float* __restrict__ l2w, const float* __restrict__ l2b,
                       float* __restrict__ out) {
    __shared__ float pooled[TOPK][66];   // 65 + pad
    __shared__ float h1[16][TOPK];
    __shared__ float p[16][10];
    __shared__ float fl[192];
    __shared__ float r1[128];

    int g = blockIdx.x, t = threadIdx.x, nt = blockDim.x;

    // gather pooled features [20,65] = [x1 | x2 | x3]
    for (int idx = t; idx < TOPK * 65; idx += nt) {
        int k = idx / 65, c = idx % 65;
        int node = topk[g * TOPK + k];
        float val;
        if (c < 32)      val = x1[(size_t)node * HID + c];
        else if (c < 64) val = x2[(size_t)node * HID + (c - 32)];
        else             val = x3[node];
        pooled[k][c] = val;
    }
    __syncthreads();

    // conv1: per-sorted-node matmul with 65-wide kernel
    for (int idx = t; idx < 16 * TOPK; idx += nt) {
        int o = idx / TOPK, k = idx % TOPK;
        float s = c1b[o];
        for (int c = 0; c < 65; ++c) s += pooled[k][c] * c1w[o * 65 + c];
        h1[o][k] = fmaxf(s, 0.0f);
    }
    __syncthreads();

    // maxpool(2,2)
    for (int idx = t; idx < 16 * 10; idx += nt) {
        int o = idx / 10, k = idx % 10;
        p[o][k] = fmaxf(h1[o][2 * k], h1[o][2 * k + 1]);
    }
    __syncthreads();

    // conv2: Conv1d(16->32, k=5, valid) -> [32,6], relu, flatten row-major
    for (int idx = t; idx < 32 * 6; idx += nt) {
        int o = idx / 6, tt = idx % 6;
        float s = c2b[o];
        for (int i = 0; i < 16; ++i)
#pragma unroll
            for (int r = 0; r < 5; ++r)
                s += p[i][tt + r] * c2w[o * 80 + i * 5 + r];
        fl[o * 6 + tt] = fmaxf(s, 0.0f);
    }
    __syncthreads();

    // lin1: 192 -> 128, relu
    for (int j = t; j < 128; j += nt) {
        float s = l1b[j];
        for (int i = 0; i < 192; ++i) s += fl[i] * l1w[i * 128 + j];
        r1[j] = fmaxf(s, 0.0f);
    }
    __syncthreads();

    // lin2: 128 -> 2
    for (int j = t; j < 2; j += nt) {
        float s = l2b[j];
#pragma unroll
        for (int i = 0; i < 128; ++i) s += r1[i] * l2w[i * 2 + j];
        out[g * 2 + j] = s;
    }
}

// ---------------------------------------------------------------------------
// Launch
// ---------------------------------------------------------------------------
static inline int cdiv(int a, int b) { return (a + b - 1) / b; }

extern "C" void kernel_launch(void* const* d_in, const int* in_sizes, int n_in,
                              void* d_out, int out_size, void* d_ws, size_t ws_size,
                              hipStream_t stream) {
    const float* x   = (const float*)d_in[0];
    const int*   ei  = (const int*)  d_in[1];
    const float* W0  = (const float*)d_in[3];
    const float* b0  = (const float*)d_in[4];
    const float* W1  = (const float*)d_in[5];
    const float* b1  = (const float*)d_in[6];
    const float* W2  = (const float*)d_in[7];
    const float* b2  = (const float*)d_in[8];
    const float* c1w = (const float*)d_in[9];
    const float* c1b = (const float*)d_in[10];
    const float* c2w = (const float*)d_in[11];
    const float* c2b = (const float*)d_in[12];
    const float* l1w = (const float*)d_in[13];
    const float* l1b = (const float*)d_in[14];
    const float* l2w = (const float*)d_in[15];
    const float* l2b = (const float*)d_in[16];
    float* out = (float*)d_out;

    const int NN = in_sizes[0] / 128;   // total nodes
    const int E  = in_sizes[1] / 2;     // total edges
    const int G  = NN / NPG;            // graphs
    const int* src = ei;
    const int* dst = ei + E;

    // workspace layout (floats)
    float* ws   = (float*)d_ws;
    float* dis  = ws;                           // NN   (also deg accumulator)
    float* bufT = dis  + NN;                    // NN*32 (XW scratch)
    float* x1   = bufT + (size_t)NN * HID;      // NN*32
    float* x2   = x1   + (size_t)NN * HID;      // NN*32
    float* x3   = x2   + (size_t)NN * HID;      // NN
    float* h3   = x3   + NN;                    // NN
    int*   topk = (int*)(h3 + NN);              // G*20

    const int nTiles = NN / 16;

    // degree / normalization
    k_zero<<<cdiv(NN, 256), 256, 0, stream>>>(dis, NN);
    k_deg<<<cdiv(E, 256), 256, 0, stream>>>(dst, dis, E);
    k_dis<<<cdiv(NN, 256), 256, 0, stream>>>(dis, NN);

    // ---- GCN layer 1: x1 = tanh(prop(x @ W0) + b0)
    k_gemm_wmma_n32<<<cdiv(nTiles, 4), 128, 0, stream>>>(x, W0, bufT, 128, nTiles);
    k_init_self32<<<cdiv(NN * HID, 256), 256, 0, stream>>>(bufT, dis, x1, NN);
    k_edge_scatter32<<<cdiv(E, 8), 256, 0, stream>>>(src, dst, bufT, dis, x1, E);
    k_fin_tanh32<<<cdiv(NN * HID, 256), 256, 0, stream>>>(x1, b0, NN);

    // ---- GCN layer 2: x2 = tanh(prop(x1 @ W1) + b1)
    k_gemm_wmma_n32<<<cdiv(nTiles, 4), 128, 0, stream>>>(x1, W1, bufT, 32, nTiles);
    k_init_self32<<<cdiv(NN * HID, 256), 256, 0, stream>>>(bufT, dis, x2, NN);
    k_edge_scatter32<<<cdiv(E, 8), 256, 0, stream>>>(src, dst, bufT, dis, x2, E);
    k_fin_tanh32<<<cdiv(NN * HID, 256), 256, 0, stream>>>(x2, b1, NN);

    // ---- GCN layer 3: x3 = tanh(prop(x2 @ W2) + b2)   (32 -> 1)
    k_node_dot32<<<cdiv(NN, 256), 256, 0, stream>>>(x2, W2, h3, NN);
    k_init_self1<<<cdiv(NN, 256), 256, 0, stream>>>(h3, dis, x3, NN);
    k_edge_scatter1<<<cdiv(E, 256), 256, 0, stream>>>(src, dst, h3, dis, x3, E);
    k_fin_tanh1<<<cdiv(NN, 256), 256, 0, stream>>>(x3, b2, NN);

    // ---- sort pool + CNN/MLP tail
    k_sortpool<<<G, NPG, 0, stream>>>(x3, topk);
    k_tail<<<G, 128, 0, stream>>>(x1, x2, x3, topk,
                                  c1w, c1b, c2w, c2b, l1w, l1b, l2w, l2b, out);
}
